// MultiHeadAttention_61572651155627
// MI455X (gfx1250) — matrix operand order repaired and checked by reading, not compiled, not run on hardware
//
#include <hip/hip_runtime.h>

// ---------------------------------------------------------------------------
// MHA for MI455X (gfx1250): bf16 WMMA everywhere, flash-attention structure.
// BS=4, SL=2048, D=1024, H=16, HD=64.
// Attention computed transposed (S^T = K*Q^T, O^T = V^T*P^T) so softmax needs
// only one shfl_xor(16) per reduction and P^T feeds WMMA straight from
// registers -- no LDS, no barriers in the attention inner loop.
// ---------------------------------------------------------------------------

typedef __bf16 bf16;
typedef __attribute__((ext_vector_type(8)))  __bf16 v8bf;
typedef __attribute__((ext_vector_type(16))) __bf16 v16bf;
typedef __attribute__((ext_vector_type(8)))  float  v8f;

__device__ __forceinline__ v8f wmma_bf16(v16bf a, v16bf b, v8f c) {
  // (neg_a, A, neg_b, B, c_mod, C, reuse_a, reuse_b)
  return __builtin_amdgcn_wmma_f32_16x16x32_bf16(false, a, false, b, (short)0, c,
                                                 false, false);
}

// A-fragment (16x32, MxK): lane holds row (lane&15); 16 contiguous K per
// lane, lane-half selects K base +16 (matches B-fragment order).
__device__ __forceinline__ v16bf load_frag_a(const bf16* p0, int ld) {
  int lane = threadIdx.x & 31;
  const bf16* p = p0 + (lane & 15) * ld + ((lane >> 4) << 4);
  union { v16bf v; v8bf h[2]; } u;
  u.h[0] = *(const v8bf*)p;
  u.h[1] = *(const v8bf*)(p + 8);
  return u.v;
}

// B-fragment (32x16, KxN): lane holds column (lane&15); 16 contiguous K per
// lane, lane-half selects K base +16 (matches A-fragment order).
__device__ __forceinline__ v16bf load_frag_b(const bf16* p0, int ld) {
  int lane = threadIdx.x & 31;
  const bf16* p = p0 + (lane & 15) * ld + ((lane >> 4) << 4);
  union { v16bf v; v8bf h[2]; } u;
  u.h[0] = *(const v8bf*)p;
  u.h[1] = *(const v8bf*)(p + 8);
  return u.v;
}

__device__ __forceinline__ unsigned pack_bf16(float x, float y) {
  union { bf16 b[2]; unsigned u; } p;
  p.b[0] = (bf16)x; p.b[1] = (bf16)y;
  return p.u;
}

// ---------------------------------------------------------------------------
// f32 -> bf16 conversion (vectorized)
// ---------------------------------------------------------------------------
__global__ __launch_bounds__(256) void cvt_f32_bf16(const float* __restrict__ s,
                                                    bf16* __restrict__ d, int n) {
  int i = (blockIdx.x * 256 + threadIdx.x) * 4;
  if (i >= n) return;
  float4 f = *(const float4*)(s + i);
  union { bf16 b[4]; unsigned long long u; } o;
  o.b[0] = (bf16)f.x; o.b[1] = (bf16)f.y; o.b[2] = (bf16)f.z; o.b[3] = (bf16)f.w;
  *(unsigned long long*)(d + i) = o.u;
}

// ---------------------------------------------------------------------------
// GEMM: C[M=8192, N=1024] = A[M,1024](bf16) * W[1024,1024](bf16, [out,in])^T + bias
// One wave -> 16x64 strip. MODE 0: bf16 out [b,h,s,d]; MODE 1: bf16 out
// [b,h,d,s] (transposed, for V); MODE 2: f32 row-major out.
// ---------------------------------------------------------------------------
template <int MODE>
__global__ __launch_bounds__(256) void gemm_bias(const bf16* __restrict__ A,
                                                 const bf16* __restrict__ W,
                                                 const float* __restrict__ bias,
                                                 bf16* __restrict__ outb,
                                                 float* __restrict__ outf) {
  const int wave = threadIdx.x >> 5;
  const int lane = threadIdx.x & 31;
  const int m0 = blockIdx.y * 128 + wave * 16;
  const int n0 = blockIdx.x * 64;

  v8f acc[4] = {};
  for (int k0 = 0; k0 < 1024; k0 += 32) {
    v16bf a = load_frag_a(A + m0 * 1024 + k0, 1024);
#pragma unroll
    for (int t = 0; t < 4; ++t) {
      v16bf b = load_frag_b(W + (n0 + t * 16) * 1024 + k0, 1024);
      acc[t] = wmma_bf16(a, b, acc[t]);
    }
  }

  const int nlo = lane & 15;
  const int mhi = (lane >> 4) << 3;
#pragma unroll
  for (int t = 0; t < 4; ++t) {
    const int n = n0 + t * 16 + nlo;
    const float bs = bias[n];
#pragma unroll
    for (int j = 0; j < 8; ++j) {
      const int m = m0 + j + mhi;
      const float v = acc[t][j] + bs;
      if (MODE == 2) {
        outf[m * 1024 + n] = v;
      } else {
        const int b = m >> 11, s = m & 2047;   // m = b*2048 + s
        const int h = n >> 6,  d = n & 63;     // n = h*64 + d
        if (MODE == 0) outb[(((b * 16 + h) * 2048) + s) * 64 + d] = (bf16)v;
        else           outb[(((b * 16 + h) * 64) + d) * 2048 + s] = (bf16)v;
      }
    }
  }
}

// ---------------------------------------------------------------------------
// Flash attention (transposed): one wave = 16 query rows of one (b,h).
// S^T = K * Q^T  -> C-layout: lane owns query column q = lane&15; lane-half
// owns key rows {r(+8)} of each 16-key tile.
// O^T = V^T * P^T accumulated over 32 keys/iter; P^T B-fragment assembled in
// registers via packed shfl_xor(16).
// Q,K: bf16 [b,h,s,64]; Vt: bf16 [b,h,64,s]; AO: bf16 [b,s,1024].
// ---------------------------------------------------------------------------
__global__ __launch_bounds__(256) void flash_attn(const bf16* __restrict__ Q,
                                                  const bf16* __restrict__ Kb,
                                                  const bf16* __restrict__ Vt,
                                                  bf16* __restrict__ AO) {
  const int wave = threadIdx.x >> 5;
  const int lane = threadIdx.x & 31;
  const int tile = blockIdx.x * 8 + wave;   // tile = (b*16+h)*128 + qt
  const int qt = tile & 127;
  const int bh = tile >> 7;                 // b*16 + h
  const int q0 = qt * 16;

  const bf16* Qp = Q  + (size_t)bh * (2048 * 64);
  const bf16* Kp = Kb + (size_t)bh * (2048 * 64);
  const bf16* Vp = Vt + (size_t)bh * (64 * 2048);

  // Q as B-fragment: column q = lane&15 is row (q0 + q) of Q (contiguous in d)
  const v16bf qb0 = load_frag_b(Qp + q0 * 64, 64);       // d = 0..31
  const v16bf qb1 = load_frag_b(Qp + q0 * 64 + 32, 64);  // d = 32..63

  float mrun = -1e30f, lrun = 0.0f;
  v8f O[4] = {};   // O^T: tile t covers d = t*16..t*16+15 (M), q (N)
  const bool lolane = (lane < 16);

  for (int key0 = 0; key0 < 2048; key0 += 32) {
    // prefetch next K tile (wave-uniform address; wraps harmlessly)
    __builtin_prefetch(Kp + ((key0 + 32) & 2047) * 64, 0, 1);

    // S^T tiles: s0 = keys key0..key0+15, s1 = keys key0+16..key0+31
    v8f s0 = {}, s1 = {};
    s0 = wmma_bf16(load_frag_a(Kp + key0 * 64, 64),            qb0, s0);
    s0 = wmma_bf16(load_frag_a(Kp + key0 * 64 + 32, 64),       qb1, s0);
    s1 = wmma_bf16(load_frag_a(Kp + (key0 + 16) * 64, 64),     qb0, s1);
    s1 = wmma_bf16(load_frag_a(Kp + (key0 + 16) * 64 + 32, 64), qb1, s1);

    // per-lane: 16 logits of query q (keys split with xor-16 partner lane)
    float t0[8], t1[8];
    float mx = -1e30f;
#pragma unroll
    for (int r = 0; r < 8; ++r) {
      t0[r] = s0[r] * 0.125f;   // 1/sqrt(64)
      t1[r] = s1[r] * 0.125f;
      mx = fmaxf(mx, fmaxf(t0[r], t1[r]));
    }
    mx = fmaxf(mx, __shfl_xor(mx, 16, 32));     // full 32-key max for this q
    const float mn = fmaxf(mrun, mx);
    const float alpha = __expf(mrun - mn);
    mrun = mn;

    float rs = 0.0f;
#pragma unroll
    for (int r = 0; r < 8; ++r) {
      t0[r] = __expf(t0[r] - mn);
      t1[r] = __expf(t1[r] - mn);
      rs += t0[r] + t1[r];
    }
    rs += __shfl_xor(rs, 16, 32);
    lrun = lrun * alpha + rs;

    // Build P^T B-fragment (32 keys x 16 q). B layout: lane<16 holds keys
    // 0..15 of column q, lane>=16 holds keys 16..31. C-layout gave us the
    // complementary halves, so swap packed pairs with one xor-16 shuffle.
    union { v16bf v; unsigned u[8]; } pb;
#pragma unroll
    for (int j = 0; j < 4; ++j) {
      const unsigned pk0 = pack_bf16(t0[2 * j], t0[2 * j + 1]);
      const unsigned pk1 = pack_bf16(t1[2 * j], t1[2 * j + 1]);
      const unsigned sh0 = __shfl_xor(pk0, 16, 32);
      const unsigned sh1 = __shfl_xor(pk1, 16, 32);
      pb.u[j]     = lolane ? pk0 : sh1;   // keys {0..7} | {16..23}
      pb.u[4 + j] = lolane ? sh0 : pk1;   // keys {8..15} | {24..31}
    }

    // O^T update: rescale by alpha (per-lane scalar!), accumulate V^T * P^T
#pragma unroll
    for (int t = 0; t < 4; ++t) {
#pragma unroll
      for (int r = 0; r < 8; ++r) O[t][r] *= alpha;
      O[t] = wmma_bf16(load_frag_a(Vp + (t * 16) * 2048 + key0, 2048), pb.v, O[t]);
    }
  }

  // Epilogue: O^T element r of tile t -> d = t*16 + r + (lane>=16 ? 8 : 0),
  // query row s = q0 + (lane&15). Packed 2xbf16 stores along d.
  const float inv = 1.0f / lrun;
  const int b = bh >> 4, h = bh & 15;
  const int s = q0 + (lane & 15);
  const int dhi = (lane >> 4) << 3;
  bf16* row = AO + ((size_t)(b * 2048 + s)) * 1024 + h * 64 + dhi;
#pragma unroll
  for (int t = 0; t < 4; ++t) {
#pragma unroll
    for (int j = 0; j < 4; ++j) {
      const unsigned pk = pack_bf16(O[t][2 * j] * inv, O[t][2 * j + 1] * inv);
      *(unsigned*)(row + t * 16 + 2 * j) = pk;
    }
  }
}

// ---------------------------------------------------------------------------
// Host launcher
// ---------------------------------------------------------------------------
extern "C" void kernel_launch(void* const* d_in, const int* in_sizes, int n_in,
                              void* d_out, int out_size, void* d_ws, size_t ws_size,
                              hipStream_t stream) {
  (void)in_sizes; (void)n_in; (void)out_size; (void)ws_size;
  const float* value  = (const float*)d_in[0];
  const float* key_in = (const float*)d_in[1];
  const float* query  = (const float*)d_in[2];
  const float* Wq = (const float*)d_in[3];  const float* bq = (const float*)d_in[4];
  const float* Wk = (const float*)d_in[5];  const float* bk = (const float*)d_in[6];
  const float* Wv = (const float*)d_in[7];  const float* bv = (const float*)d_in[8];
  const float* Wo = (const float*)d_in[9];  const float* bo = (const float*)d_in[10];
  float* out = (float*)d_out;

  char* ws = (char*)d_ws;
  const size_t MB = (size_t)1 << 20;
  bf16* XQ  = (bf16*)(ws +   0 * MB);   // 16 MB each activation (8192x1024 bf16)
  bf16* XK  = (bf16*)(ws +  16 * MB);
  bf16* XV  = (bf16*)(ws +  32 * MB);
  bf16* WQb = (bf16*)(ws +  48 * MB);   // 2 MB each weight
  bf16* WKb = (bf16*)(ws +  50 * MB);
  bf16* WVb = (bf16*)(ws +  52 * MB);
  bf16* WOb = (bf16*)(ws +  54 * MB);
  bf16* Qb  = (bf16*)(ws +  56 * MB);   // [b,h,s,d]
  bf16* Kbf = (bf16*)(ws +  72 * MB);   // [b,h,s,d]
  bf16* Vt  = (bf16*)(ws +  88 * MB);   // [b,h,d,s]
  bf16* AO  = (bf16*)(ws + 104 * MB);   // [b,s,D]

  const int NACT = 8192 * 1024;
  const int NW   = 1024 * 1024;

  cvt_f32_bf16<<<NACT / 1024, 256, 0, stream>>>(query,  XQ, NACT);
  cvt_f32_bf16<<<NACT / 1024, 256, 0, stream>>>(key_in, XK, NACT);
  cvt_f32_bf16<<<NACT / 1024, 256, 0, stream>>>(value,  XV, NACT);
  cvt_f32_bf16<<<NW   / 1024, 256, 0, stream>>>(Wq, WQb, NW);
  cvt_f32_bf16<<<NW   / 1024, 256, 0, stream>>>(Wk, WKb, NW);
  cvt_f32_bf16<<<NW   / 1024, 256, 0, stream>>>(Wv, WVb, NW);
  cvt_f32_bf16<<<NW   / 1024, 256, 0, stream>>>(Wo, WOb, NW);

  dim3 ggrid(16, 64);  // N/64, M/128
  gemm_bias<0><<<ggrid, 256, 0, stream>>>(XQ, WQb, bq, Qb,  nullptr);
  gemm_bias<0><<<ggrid, 256, 0, stream>>>(XK, WKb, bk, Kbf, nullptr);
  gemm_bias<1><<<ggrid, 256, 0, stream>>>(XV, WVb, bv, Vt,  nullptr);

  flash_attn<<<1024, 256, 0, stream>>>(Qb, Kbf, Vt, AO);

  gemm_bias<2><<<ggrid, 256, 0, stream>>>(AO, WOb, bo, nullptr, out);
}
